// GraphSAGE_67456756351010
// MI455X (gfx1250) — matrix operand order repaired and checked by reading, not compiled, not run on hardware
//
#include <hip/hip_runtime.h>
#include <hip/hip_bf16.h>

// ---- vector types for WMMA ----
typedef __attribute__((ext_vector_type(16))) _Float16 v16h;
typedef __attribute__((ext_vector_type(8)))  _Float16 v8h;
typedef __attribute__((ext_vector_type(8)))  float    v8f;

#define N_NODES 50000
#define N_EDGES 640000
#define D 128

// ---------------------------------------------------------------------------
// elementwise f32 -> f16 convert
__global__ void sage_cvt_f16(const float* __restrict__ in,
                             _Float16* __restrict__ out, int n) {
    int i = blockIdx.x * blockDim.x + threadIdx.x;
    int stride = gridDim.x * blockDim.x;
    for (; i < n; i += stride) out[i] = (_Float16)in[i];
}

// zero a f32 buffer
__global__ void sage_zero(float* __restrict__ p, int n) {
    int i = blockIdx.x * blockDim.x + threadIdx.x;
    int stride = gridDim.x * blockDim.x;
    for (; i < n; i += stride) p[i] = 0.0f;
}

// ---------------------------------------------------------------------------
// scatter-mean numerator: one wave32 per edge, float4 per lane (32*4 = 128)
__global__ void sage_scatter(const float* __restrict__ feat,
                             const long long* __restrict__ src,
                             const long long* __restrict__ dst,
                             float* __restrict__ agg,
                             float* __restrict__ cnt, int nEdges) {
    int edge = (int)((blockIdx.x * (unsigned)blockDim.x + threadIdx.x) >> 5);
    int lane = threadIdx.x & 31;
    if (edge >= nEdges) return;
    long long s = src[edge];
    long long d = dst[edge];
    const float4 v = ((const float4*)(feat + s * D))[lane];
    float* ap = agg + d * D + lane * 4;
    atomicAdd(ap + 0, v.x);
    atomicAdd(ap + 1, v.y);
    atomicAdd(ap + 2, v.z);
    atomicAdd(ap + 3, v.w);
    if (lane == 0) atomicAdd(cnt + d, 1.0f);
}

// mean = agg / max(cnt,1)  -> f16
__global__ void sage_mean(const float* __restrict__ agg,
                          const float* __restrict__ cnt,
                          _Float16* __restrict__ meanh) {
    int i = blockIdx.x * blockDim.x + threadIdx.x;
    int stride = gridDim.x * blockDim.x;
    int n = N_NODES * D;
    for (; i < n; i += stride) {
        float c = cnt[i / D];
        float m = agg[i] / fmaxf(c, 1.0f);
        meanh[i] = (_Float16)m;
    }
}

// ---------------------------------------------------------------------------
// out[r,c] = sum_k A1[r,k]*Wl[c,k] + bias[c] + sum_k A2[r,k]*Wr[c,k]
// One wave per 16x16 output tile. Block = 256 threads = 8 waves = 8 col tiles.
__global__ __launch_bounds__(256) void sage_gemm(
    const _Float16* __restrict__ A1,   // mean  [N,128] f16
    const _Float16* __restrict__ A2,   // root  [N,128] f16
    const _Float16* __restrict__ Wl,   // [128,128] f16 row-major (out,k)
    const _Float16* __restrict__ Wr,   // [128,128] f16
    const float* __restrict__ bias,    // [128]
    float* __restrict__ out32,         // [N,128] f32
    _Float16* __restrict__ out16,      // optional [N,128] f16 (may be null)
    int relu) {
    const int row0 = blockIdx.x * 16;
    const int col0 = (threadIdx.x >> 5) * 16;
    const int lane = threadIdx.x & 31;
    const int m16  = lane & 15;
    const int hi   = (lane & 16) ? 1 : 0;

    // A layout (16-bit A 16x32): lane row = lane%16; lanes<16 hold K 0..15,
    // lanes>=16 hold K 16..31 (matching B)  -> two 8-half contiguous loads per chunk.
    const int klo = hi ? 16 : 0;
    // B layout (16-bit B 32x16): column N = lane%16; lanes<16 hold K 0..15,
    // lanes>=16 hold K 16..31, in K order -> one 16-half contiguous load.
    const int kboff = hi ? 16 : 0;

    const _Float16* a1row = A1 + (size_t)(row0 + m16) * D;
    const _Float16* a2row = A2 + (size_t)(row0 + m16) * D;
    const _Float16* wlcol = Wl + (size_t)(col0 + m16) * D;  // row of W == column of B
    const _Float16* wrcol = Wr + (size_t)(col0 + m16) * D;

    v8f acc = {};
#pragma unroll
    for (int kk = 0; kk < D; kk += 32) {
        v8h a1lo = *(const v8h*)(a1row + kk + klo);
        v8h a1hi = *(const v8h*)(a1row + kk + 8 + klo);
        v16h a1 = __builtin_shufflevector(a1lo, a1hi,
                    0,1,2,3,4,5,6,7, 8,9,10,11,12,13,14,15);
        v16h b1 = *(const v16h*)(wlcol + kk + kboff);
        acc = __builtin_amdgcn_wmma_f32_16x16x32_f16(
                  false, a1, false, b1, (short)0, acc, false, false);

        v8h a2lo = *(const v8h*)(a2row + kk + klo);
        v8h a2hi = *(const v8h*)(a2row + kk + 8 + klo);
        v16h a2 = __builtin_shufflevector(a2lo, a2hi,
                    0,1,2,3,4,5,6,7, 8,9,10,11,12,13,14,15);
        v16h b2 = *(const v16h*)(wrcol + kk + kboff);
        acc = __builtin_amdgcn_wmma_f32_16x16x32_f16(
                  false, a2, false, b2, (short)0, acc, false, false);
    }

    // C/D layout: VGPR i -> row = i + 8*hi, col = lane%16
    const float bc = bias[col0 + m16];
    const int col = col0 + m16;
#pragma unroll
    for (int i = 0; i < 8; ++i) {
        float v = acc[i] + bc;
        if (relu) v = fmaxf(v, 0.0f);
        const size_t idx = (size_t)(row0 + i + 8 * hi) * D + col;
        out32[idx] = v;
        if (out16) out16[idx] = (_Float16)v;
    }
}

// ---------------------------------------------------------------------------
extern "C" void kernel_launch(void* const* d_in, const int* in_sizes, int n_in,
                              void* d_out, int out_size, void* d_ws, size_t ws_size,
                              hipStream_t stream) {
    const float*     x     = (const float*)d_in[0];
    const long long* edges = (const long long*)d_in[1];   // int64 [2, E]
    const float*     W1l   = (const float*)d_in[2];
    const float*     b1l   = (const float*)d_in[3];
    const float*     W1r   = (const float*)d_in[4];
    const float*     W2l   = (const float*)d_in[5];
    const float*     b2l   = (const float*)d_in[6];
    const float*     W2r   = (const float*)d_in[7];
    float* out = (float*)d_out;

    const long long* src = edges;
    const long long* dst = edges + N_EDGES;

    // ---- workspace carve-up (256B aligned), ~90 MB total ----
    char* ws = (char*)d_ws;
    size_t off = 0;
    auto carve = [&](size_t bytes) {
        void* p = ws + off;
        off += (bytes + 255) & ~(size_t)255;
        return p;
    };
    float*     agg   = (float*)    carve((size_t)N_NODES * D * 4);
    float*     cnt   = (float*)    carve((size_t)N_NODES * 4);
    _Float16*  xh    = (_Float16*) carve((size_t)N_NODES * D * 2);
    _Float16*  meanh = (_Float16*) carve((size_t)N_NODES * D * 2);
    float*     h32   = (float*)    carve((size_t)N_NODES * D * 4);
    _Float16*  h16   = (_Float16*) carve((size_t)N_NODES * D * 2);
    _Float16*  w1lh  = (_Float16*) carve((size_t)D * D * 2);
    _Float16*  w1rh  = (_Float16*) carve((size_t)D * D * 2);
    _Float16*  w2lh  = (_Float16*) carve((size_t)D * D * 2);
    _Float16*  w2rh  = (_Float16*) carve((size_t)D * D * 2);

    const int TB = 256;
    const int nFeat = N_NODES * D;
    const int nW = D * D;

    sage_cvt_f16<<<(nW + TB - 1) / TB, TB, 0, stream>>>(W1l, w1lh, nW);
    sage_cvt_f16<<<(nW + TB - 1) / TB, TB, 0, stream>>>(W1r, w1rh, nW);
    sage_cvt_f16<<<(nW + TB - 1) / TB, TB, 0, stream>>>(W2l, w2lh, nW);
    sage_cvt_f16<<<(nW + TB - 1) / TB, TB, 0, stream>>>(W2r, w2rh, nW);
    sage_cvt_f16<<<4096, TB, 0, stream>>>(x, xh, nFeat);

    const int scatterBlocks = (N_EDGES * 32 + TB - 1) / TB;   // one wave32 per edge
    const int gemmBlocks = N_NODES / 16;                      // 3125 exact

    // -------- layer 1 --------
    sage_zero<<<4096, TB, 0, stream>>>(agg, nFeat);
    sage_zero<<<(N_NODES + TB - 1) / TB, TB, 0, stream>>>(cnt, N_NODES);
    sage_scatter<<<scatterBlocks, TB, 0, stream>>>(x, src, dst, agg, cnt, N_EDGES);
    sage_mean<<<4096, TB, 0, stream>>>(agg, cnt, meanh);
    sage_gemm<<<gemmBlocks, TB, 0, stream>>>(meanh, xh, w1lh, w1rh, b1l,
                                             h32, h16, /*relu=*/1);

    // -------- layer 2 --------
    sage_zero<<<4096, TB, 0, stream>>>(agg, nFeat);
    sage_zero<<<(N_NODES + TB - 1) / TB, TB, 0, stream>>>(cnt, N_NODES);
    sage_scatter<<<scatterBlocks, TB, 0, stream>>>(h32, src, dst, agg, cnt, N_EDGES);
    sage_mean<<<4096, TB, 0, stream>>>(agg, cnt, meanh);
    sage_gemm<<<gemmBlocks, TB, 0, stream>>>(meanh, h16, w2lh, w2rh, b2l,
                                             out, (_Float16*)nullptr, /*relu=*/0);

    (void)in_sizes; (void)n_in; (void)out_size; (void)ws_size;
}
